// PseudoMambaRLCore_35631048688100
// MI455X (gfx1250) — matrix operand order repaired and checked by reading, not compiled, not run on hardware
//
#include <hip/hip_runtime.h>
#include <hip/hip_bf16.h>

// ---------------------------------------------------------------------------
// PseudoMambaRLCore single-step on gfx1250 (MI455X, wave32, WMMA).
// GEMMs: tiled v_wmma_f32_16x16x32_f16, block tile 128x64, 8 waves (2Mx4N),
// 4 WMMA per wave per 32-k step, double-buffered LDS with sched_barrier-pinned
// phases: all 6 global_load_b128 issue back-to-back, WMMAs + ds_loads execute
// while they are in flight, and s_wait_loadcnt lands after the matrix math.
// Staging is branchless (clamp+select on ragged N, packed f32->f16 cvt,
// b64 LDS stores). Requires M%128==0, K%32==0 (true at every call site).
// Elementwise SSM/conv stages are fused streaming kernels (memory-bound:
// ~540MB ssm_state traffic / 23.3 TB/s ~ 23us dominates total runtime).
// ---------------------------------------------------------------------------

typedef __attribute__((ext_vector_type(16))) _Float16 v16h;
typedef __attribute__((ext_vector_type(4)))  _Float16 v4h;
typedef __attribute__((ext_vector_type(8)))  float    v8f;

#if defined(__has_builtin)
#if __has_builtin(__builtin_amdgcn_sched_barrier)
#define SCHED_FENCE() __builtin_amdgcn_sched_barrier(0)
#endif
#endif
#ifndef SCHED_FENCE
#define SCHED_FENCE()
#endif

#define B_   2048
#define OBS_ 256
#define DM_  1024
#define DI_  2048
#define DS_  16
#define DC_  4
#define DTR_ 64

#define TM 128
#define TN 64
#define TK 32
#define PADK 8   // row stride 40 halves = 80B (8/16B aligned)

__device__ __forceinline__ float softplus_f(float x) {
    return (x > 20.0f) ? x : log1pf(__expf(x));
}
__device__ __forceinline__ float silu_f(float x) {
    return x / (1.0f + __expf(-x));
}

// C[M,N] = act( A[M,K(lda)] * Bm[N,K(ldb)]^T + bias[N] ),  act: 0=id, 1=softplus
// Assumes M % TM == 0 and K % TK == 0. N may be ragged (guarded).
__global__ __launch_bounds__(256)
void gemm_wmma_f16(const float* __restrict__ A, const float* __restrict__ Bm,
                   const float* __restrict__ bias, float* __restrict__ C,
                   int M, int N, int K, int lda, int ldb, int ldc, int act)
{
    __shared__ _Float16 As[2][TM][TK + PADK];
    __shared__ _Float16 Bs[2][TN][TK + PADK];

    const int tid  = threadIdx.x;
    const int lane = tid & 31;
    const int wid  = tid >> 5;      // 8 waves
    const int wm   = wid >> 2;      // 0..1 : 64-row band (4 x 16-row tiles)
    const int wn   = wid & 3;       // 0..3 : 16-col band

    const int m0 = blockIdx.y * TM;
    const int n0 = blockIdx.x * TN;

    v8f acc[4] = {v8f{}, v8f{}, v8f{}, v8f{}};

    const int mfrag = lane & 15;          // A: row within 16x16 tile
    const int khalf = lane >> 4;          // A: low/high K half-wave
    const int nfrag = lane & 15;          // B: column
    const int kbB   = (lane >> 4) << 4;   // B: K base (0 or 16)

    // staging coords: one float4 (4 consecutive k) per thread per chunk
    const int sr  = tid >> 3;             // row  (8 float4 per 32-wide row)
    const int sc4 = (tid & 7) << 2;       // k sub-offset

    float4 ra[4], rb[2];

    auto load_regs = [&](int k0) {
        #pragma unroll
        for (int i = 0; i < 4; ++i)
            ra[i] = *(const float4*)(A + (size_t)(m0 + sr + i * 32) * lda + k0 + sc4);
        #pragma unroll
        for (int i = 0; i < 2; ++i) {
            int gn  = n0 + sr + i * 32;
            int gnc = (gn < N) ? gn : 0;
            rb[i] = *(const float4*)(Bm + (size_t)gnc * ldb + k0 + sc4);
        }
    };
    auto store_tile = [&](int buf) {
        #pragma unroll
        for (int i = 0; i < 4; ++i) {
            v4h hv;
            hv[0] = (_Float16)ra[i].x; hv[1] = (_Float16)ra[i].y;
            hv[2] = (_Float16)ra[i].z; hv[3] = (_Float16)ra[i].w;
            *(v4h*)&As[buf][sr + i * 32][sc4] = hv;
        }
        #pragma unroll
        for (int i = 0; i < 2; ++i) {
            const bool ok = (n0 + sr + i * 32) < N;
            v4h hv;
            hv[0] = ok ? (_Float16)rb[i].x : (_Float16)0.0f;
            hv[1] = ok ? (_Float16)rb[i].y : (_Float16)0.0f;
            hv[2] = ok ? (_Float16)rb[i].z : (_Float16)0.0f;
            hv[3] = ok ? (_Float16)rb[i].w : (_Float16)0.0f;
            *(v4h*)&Bs[buf][sr + i * 32][sc4] = hv;
        }
    };
    auto compute = [&](int buf) {
        // B fragment: 32x16 (KxN); lanes0-15 K=0..15, lanes16-31 K=16..31
        v16h bfrag = {};
        #pragma unroll
        for (int v = 0; v < 8; ++v) {
            bfrag[2*v]   = Bs[buf][wn*16 + nfrag][kbB + 2*v];
            bfrag[2*v+1] = Bs[buf][wn*16 + nfrag][kbB + 2*v + 1];
        }
        // 4 A fragments (16x32 MxK, lanes0-15 K=0..15, lanes16-31 K=16..31) + 4 WMMA
        #pragma unroll
        for (int t = 0; t < 4; ++t) {
            v16h af = {};
            const int rbw = wm * 64 + t * 16 + mfrag;
            #pragma unroll
            for (int v = 0; v < 8; ++v) {
                int kk = (khalf << 4) + (v << 1);
                af[2*v]   = As[buf][rbw][kk];
                af[2*v+1] = As[buf][rbw][kk + 1];
            }
            acc[t] = __builtin_amdgcn_wmma_f32_16x16x32_f16(false, af, false, bfrag,
                                                            (short)0, acc[t], false, false);
        }
    };

    // ---- software pipeline: prologue ----
    load_regs(0);
    store_tile(0);
    __syncthreads();
    int buf = 0;
    // ---- steady state: loads | fence | WMMA | fence | cvt+LDS store ----
    #pragma unroll 1
    for (int k0 = TK; k0 < K; k0 += TK) {
        load_regs(k0);          // 6 global_load_b128 issued back-to-back
        SCHED_FENCE();          // keep loads up here, results live across fence
        compute(buf);           // WMMAs overlap the in-flight loads
        SCHED_FENCE();          // first load consumer (cvt) stays below WMMAs
        store_tile(buf ^ 1);    // s_wait_loadcnt lands here
        __syncthreads();
        buf ^= 1;
    }
    compute(buf);               // pipeline epilogue tile

    // ---- epilogue: C/D layout lanes0-15 M=v, lanes16-31 M=v+8; N=lane&15 ----
    const int col   = n0 + wn * 16 + (lane & 15);
    const int rbase = (lane >> 4) * 8;
    if (col < N) {
        const float bv = bias ? bias[col] : 0.0f;
        #pragma unroll
        for (int t = 0; t < 4; ++t) {
            #pragma unroll
            for (int v = 0; v < 8; ++v) {
                int row = m0 + wm * 64 + t * 16 + rbase + v;  // always < M (M%TM==0)
                float r = acc[t][v] + bv;
                if (act == 1) r = softplus_f(r);
                C[(size_t)row * ldc + col] = r;
            }
        }
    }
}

// negA = -exp(A_log)   (batch-invariant, hoisted out of the 67M-element loop)
__global__ void negA_kernel(const float* __restrict__ A_log, float* __restrict__ negA, int n)
{
    int i = blockIdx.x * blockDim.x + threadIdx.x;
    if (i < n) negA[i] = -__expf(A_log[i]);
}

// Depthwise causal conv state roll + conv + SiLU. Writes new conv_state to
// d_out region and activated x to workspace.
__global__ __launch_bounds__(256)
void conv_silu_kernel(const float* __restrict__ xz,
                      const float* __restrict__ cs_in,
                      const float* __restrict__ conv_w,
                      const float* __restrict__ conv_b,
                      float* __restrict__ cs_out,
                      float* __restrict__ x_act)
{
    int idx = blockIdx.x * blockDim.x + threadIdx.x;
    if (idx >= B_ * DI_) return;
    int b = idx / DI_, d = idx - b * DI_;

    const float* cs = cs_in + (size_t)idx * DC_;
    float c1 = cs[1], c2 = cs[2], c3 = cs[3];
    float xn = xz[(size_t)b * (2 * DI_) + d];

    const float* w = conv_w + d * DC_;
    float s = c1 * w[0] + c2 * w[1] + c3 * w[2] + xn * w[3] + conv_b[d];
    x_act[idx] = silu_f(s);

    float4 o; o.x = c1; o.y = c2; o.z = c3; o.w = xn;
    *((float4*)(cs_out + (size_t)idx * DC_)) = o;   // new conv state (rolled)
}

// Fused SSM recurrence: state' = state*exp(dt*(-exp(A_log))) + x*dt*B,
// y = state'.C + D*x, y *= silu(z). One pass over 540MB of state traffic.
// y_buf aliases x_act: each thread reads x_act[idx] then overwrites it with y.
__global__ __launch_bounds__(256)
void ssm_kernel(const float* __restrict__ dt_in,
                const float* __restrict__ xz,
                const float* __restrict__ x_db,
                const float* __restrict__ negA,
                const float* __restrict__ Dp,
                const float* __restrict__ ss_in,
                float* __restrict__ ss_out,
                float* __restrict__ y_buf)
{
    int idx = blockIdx.x * blockDim.x + threadIdx.x;
    if (idx >= B_ * DI_) return;
    int b = idx / DI_, d = idx - b * DI_;

    float dtv = dt_in[idx];                 // softplus fused into GEMM epilogue
    float xa  = y_buf[idx];                 // x_act (aliased)
    float z   = xz[(size_t)b * (2 * DI_) + DI_ + d];

    const float* BC = x_db + (size_t)b * (DTR_ + 2 * DS_);   // [dt|B|C] row
    const float* st = ss_in + (size_t)idx * DS_;
    const float* nA = negA + d * DS_;
    const float xdt = xa * dtv;

    float out[DS_];
    float acc = 0.0f;
    #pragma unroll
    for (int n = 0; n < DS_; ++n) {
        float dA = __expf(dtv * nA[n]);
        float s  = st[n] * dA + xdt * BC[DTR_ + n];
        out[n] = s;
        acc += s * BC[DTR_ + DS_ + n];
    }
    float yv = (acc + Dp[d] * xa) * silu_f(z);

    float4* so = (float4*)(ss_out + (size_t)idx * DS_);
    #pragma unroll
    for (int q = 0; q < 4; ++q) {
        float4 v; v.x = out[4*q]; v.y = out[4*q+1]; v.z = out[4*q+2]; v.w = out[4*q+3];
        so[q] = v;
    }
    y_buf[idx] = yv;
}

extern "C" void kernel_launch(void* const* d_in, const int* in_sizes, int n_in,
                              void* d_out, int out_size, void* d_ws, size_t ws_size,
                              hipStream_t stream) {
    const float* obs    = (const float*)d_in[0];
    const float* cs_in  = (const float*)d_in[1];
    const float* ss_in  = (const float*)d_in[2];
    const float* W_obs  = (const float*)d_in[3];
    const float* b_obs  = (const float*)d_in[4];
    const float* W_in   = (const float*)d_in[5];
    const float* b_in   = (const float*)d_in[6];
    const float* conv_w = (const float*)d_in[7];
    const float* conv_b = (const float*)d_in[8];
    const float* W_x    = (const float*)d_in[9];
    const float* W_dt   = (const float*)d_in[10];
    const float* b_dt   = (const float*)d_in[11];
    const float* A_log  = (const float*)d_in[12];
    const float* Dp     = (const float*)d_in[13];
    const float* W_out  = (const float*)d_in[14];
    const float* b_out  = (const float*)d_in[15];

    float* out = (float*)d_out;
    float* cs_out = out + (size_t)B_ * DM_;                    // tuple slot 1
    float* ss_out = cs_out + (size_t)B_ * DI_ * DC_;           // tuple slot 2

    // workspace layout (floats); y aliases x_act (safe: read-then-write per idx)
    float* ws = (float*)d_ws;
    size_t off = 0;
    float* h      = ws + off; off += (size_t)B_ * DM_;          //  8 MB
    float* xz     = ws + off; off += (size_t)B_ * 2 * DI_;      // 32 MB
    float* x_act  = ws + off; off += (size_t)B_ * DI_;          // 16 MB
    float* x_db   = ws + off; off += (size_t)B_ * (DTR_ + 2 * DS_);
    float* dtb    = ws + off; off += (size_t)B_ * DI_;          // 16 MB
    float* negA   = ws + off; off += (size_t)DI_ * DS_;
    (void)off; (void)ws_size; (void)in_sizes; (void)n_in; (void)out_size;

    dim3 blk(256);
    auto grid = [](int M, int N) { return dim3((N + TN - 1) / TN, (M + TM - 1) / TM); };
    const int ne = B_ * DI_;
    const int eb = (ne + 255) / 256;

    // independent tiny precompute
    negA_kernel<<<(DI_ * DS_ + 255) / 256, blk, 0, stream>>>(A_log, negA, DI_ * DS_);

    // h = obs @ W_obs^T + b_obs
    gemm_wmma_f16<<<grid(B_, DM_), blk, 0, stream>>>(obs, W_obs, b_obs, h,
        B_, DM_, OBS_, OBS_, OBS_, DM_, 0);
    // xz = h @ W_in^T + b_in
    gemm_wmma_f16<<<grid(B_, 2 * DI_), blk, 0, stream>>>(h, W_in, b_in, xz,
        B_, 2 * DI_, DM_, DM_, DM_, 2 * DI_, 0);
    // conv roll + depthwise conv + SiLU -> x_act, new conv_state
    conv_silu_kernel<<<eb, blk, 0, stream>>>(xz, cs_in, conv_w, conv_b, cs_out, x_act);
    // x_db = x_act @ W_x^T   (N=96, guarded via clamp+select)
    gemm_wmma_f16<<<grid(B_, DTR_ + 2 * DS_), blk, 0, stream>>>(x_act, W_x, nullptr, x_db,
        B_, DTR_ + 2 * DS_, DI_, DI_, DI_, DTR_ + 2 * DS_, 0);
    // dt = softplus(x_db[:, :64] @ W_dt^T + b_dt)   (lda = 96, softplus fused)
    gemm_wmma_f16<<<grid(B_, DI_), blk, 0, stream>>>(x_db, W_dt, b_dt, dtb,
        B_, DI_, DTR_, DTR_ + 2 * DS_, DTR_, DI_, 1);
    // fused SSM recurrence + gate -> new ssm_state, y (in x_act slot)
    ssm_kernel<<<eb, blk, 0, stream>>>(dtb, xz, x_db, negA, Dp, ss_in, ss_out, x_act);
    // out = y @ W_out^T + b_out
    gemm_wmma_f16<<<grid(B_, DM_), blk, 0, stream>>>(x_act, W_out, b_out, out,
        B_, DM_, DI_, DI_, DI_, DM_, 0);
}
